// SparseFFN_40759239639957
// MI455X (gfx1250) — matrix operand order repaired and checked by reading, not compiled, not run on hardware
//
#include <hip/hip_runtime.h>
#include <hip/hip_bf16.h>

// ---------------- problem constants ----------------
#define BATCH    4
#define SEQ      4096
#define M_TOT    (BATCH * SEQ)     // 16384 rows
#define D_MODEL  1024
#define D_FF     4096
#define K_SEL    2048              // d_ff * (1 - 0.5)

// WMMA tiling
#define TILE     16
#define KSTEP    32                // bf16 16x16x32

typedef __bf16 v16bf __attribute__((ext_vector_type(16)));
typedef float  v8f   __attribute__((ext_vector_type(8)));
typedef unsigned int su4 __attribute__((ext_vector_type(4)));
typedef unsigned int su8 __attribute__((ext_vector_type(8)));

union FragBF { v16bf v; uint4 q[2]; };

// ---------------- workspace layout (bytes) ----------------
#define WS_XB    0
#define WS_W1B   (WS_XB    + (size_t)M_TOT   * D_MODEL * 2)
#define WS_WESTB (WS_W1B   + (size_t)D_FF    * D_MODEL * 2)
#define WS_W2B   (WS_WESTB + (size_t)D_FF    * D_MODEL * 2)
#define WS_INTER (WS_W2B   + (size_t)D_MODEL * D_FF    * 2)
#define WS_SPART (WS_INTER + (size_t)M_TOT   * K_SEL   * 2)
#define WS_SCORE (WS_SPART + (size_t)16      * D_FF    * 4)
#define WS_SEL   (WS_SCORE + (size_t)D_FF    * 4)

// ---------------- WMMA helpers ----------------
__device__ inline v8f wmma_bf16(v16bf a, v16bf b, v8f c) {
    return __builtin_amdgcn_wmma_f32_16x16x32_bf16(
        false, a, false, b, (short)0, c, false, false);
}

// A fragment (16x32 bf16, row-major, leading dim ld). ISA 7.12.2 layout.
__device__ inline v16bf load_a_frag(const __bf16* base, int ld, int m0,
                                    int ks, int lane) {
    const int m  = m0 + (lane & 15);
    const int k0 = ks + ((lane >> 4) << 3);
    const __bf16* p = base + (size_t)m * ld + k0;
    FragBF f;
    f.q[0] = *(const uint4*)(p);
    f.q[1] = *(const uint4*)(p + 16);
    return f.v;
}

// B fragment (32x16 bf16); column n contiguous along K.
__device__ inline v16bf load_b_frag(const __bf16* base, int ldk, int ks,
                                    int lane) {
    const int n  = lane & 15;
    const int k0 = ks + ((lane >> 4) << 3);
    const __bf16* p = base + (size_t)n * ldk + k0;
    FragBF f;
    f.q[0] = *(const uint4*)(p);
    f.q[1] = *(const uint4*)(p + 16);
    return f.v;
}

// ---------------- Tensor Data Mover (ISA 08_async_tensor §8) ----------------
// 1-D contiguous tile: `elems` bf16 elements from gptr -> LDS[lds_off].
__device__ inline void tdm_load_1d(unsigned int lds_off, const void* gptr,
                                   unsigned int elems) {
    su4 g0; su8 g1;
    unsigned long long ga = (unsigned long long)gptr;
    g0[0] = 1u;                                       // count=1, no gather
    g0[1] = lds_off;                                  // D#.lds_addr
    g0[2] = (unsigned int)ga;                         // global_addr[31:0]
    g0[3] = (unsigned int)((ga >> 32) & 0x01FFFFFFu)  // global_addr[56:32]
          | (2u << 30);                               // type=2 (image)
    g1[0] = (1u << 16);                               // data_size=1 -> 2 bytes
    g1[1] = (elems & 0xFFFFu) << 16;                  // tensor_dim0[15:0]
    g1[2] = (elems >> 16) | (1u << 16);               // tensor_dim0_hi | tensor_dim1=1
    g1[3] = (elems & 0xFFFFu) << 16;                  // tile_dim0 = elems (<=65535)
    g1[4] = 0u;                                       // tile_dim1/2 unused
    g1[5] = elems;                                    // tensor_dim0_stride lo
    g1[6] = 0u; g1[7] = 0u;
    asm volatile("tensor_load_to_lds %0, %1" :: "s"(g0), "s"(g1) : "memory");
}

// Gather-mode tile: 16 rows of `row_elems` bf16, row r taken at idx[r], from a
// [nrows, row_elems] row-major tensor at gbase -> LDS[lds_off], rows compacted.
__device__ inline void tdm_gather_rows16(unsigned int lds_off,
                                         const __bf16* gbase,
                                         unsigned int row_elems,
                                         unsigned int nrows,
                                         const int* __restrict__ idx) {
    su4 g0; su8 g1; su4 g2; su4 g3;
    unsigned long long ga = (unsigned long long)gbase;
    g0[0] = 1u | (1u << 31);                          // count=1, gather_mode, 16b idx
    g0[1] = lds_off;
    g0[2] = (unsigned int)ga;
    g0[3] = (unsigned int)((ga >> 32) & 0x01FFFFFFu) | (2u << 30);
    g1[0] = (1u << 16);                               // data_size = 2 bytes
    g1[1] = (row_elems & 0xFFFFu) << 16;              // tensor_dim0 lo
    g1[2] = (row_elems >> 16) | (nrows << 16);        // dim0 hi | tensor_dim1 lo
    g1[3] = (nrows >> 16) | ((row_elems & 0xFFFFu) << 16); // dim1 hi | tile_dim0
    g1[4] = 16u;                                      // tile_dim1 = #valid indices
    g1[5] = row_elems;                                // tensor_dim0_stride (elems)
    g1[6] = 0u; g1[7] = 0u;
#pragma unroll
    for (int i = 0; i < 4; ++i)
        g2[i] = ((unsigned int)idx[2 * i] & 0xFFFFu) |
                (((unsigned int)idx[2 * i + 1] & 0xFFFFu) << 16);
#pragma unroll
    for (int i = 0; i < 4; ++i)
        g3[i] = ((unsigned int)idx[8 + 2 * i] & 0xFFFFu) |
                (((unsigned int)idx[8 + 2 * i + 1] & 0xFFFFu) << 16);
    asm volatile("tensor_load_to_lds %0, %1, %2, %3"
                 :: "s"(g0), "s"(g1), "s"(g2), "s"(g3) : "memory");
}

// ---------------- kernels ----------------

__global__ void cvt_f32_bf16(const float* __restrict__ in,
                             __bf16* __restrict__ out, int n4) {
    int i = (blockIdx.x * blockDim.x + threadIdx.x);
    if (i < n4) {
        float4 v = ((const float4*)in)[i];
        __bf16* o = out + (size_t)i * 4;
        o[0] = (__bf16)v.x; o[1] = (__bf16)v.y;
        o[2] = (__bf16)v.z; o[3] = (__bf16)v.w;
    }
}

// scores partials: TDM-staged w_est tile; 8 waves x 8 M-tiles (2 groups of 4)
__global__ __launch_bounds__(256)
void scores_gemm(const __bf16* __restrict__ xb,
                 const __bf16* __restrict__ westb,
                 const float*  __restrict__ b_est,
                 float* __restrict__ spart) {
    __shared__ __align__(16) __bf16 Bs[TILE * D_MODEL];   // 32KB
    __shared__ float red[256];

    const int tid   = threadIdx.x;
    const int ntile = blockIdx.x;           // 0..255
    const int mchk  = blockIdx.y;           // 0..15
    const int n0    = ntile * TILE;

    // rows [n0..n0+15] of w_est are contiguous: one 1-D TDM transfer, wave 0.
    if (tid < 32) {
        tdm_load_1d((unsigned int)(uintptr_t)&Bs[0],
                    westb + (size_t)n0 * D_MODEL, TILE * D_MODEL);
        __builtin_amdgcn_s_wait_tensorcnt(0);
    }
    __syncthreads();

    const int wave = tid >> 5, lane = tid & 31;
    const float bias = b_est[n0 + (lane & 15)];
    float colsum = 0.0f;

    for (int g = 0; g < 2; ++g) {
        const int mt = mchk * 64 + wave * 8 + g * 4;
        v8f acc[4] = {};
        for (int ks = 0; ks < D_MODEL; ks += KSTEP) {
            v16bf b = load_b_frag(Bs, D_MODEL, ks, lane);
            __builtin_prefetch(xb + (size_t)(mt * TILE) * D_MODEL + ks + 256, 0, 1);
#pragma unroll
            for (int i = 0; i < 4; ++i) {
                v16bf a = load_a_frag(xb, D_MODEL, (mt + i) * TILE, ks, lane);
                acc[i] = wmma_bf16(a, b, acc[i]);
            }
        }
#pragma unroll
        for (int i = 0; i < 4; ++i)
#pragma unroll
            for (int v = 0; v < 8; ++v)
                colsum += fabsf(acc[i][v] + bias);
    }

    red[tid] = colsum;
    __syncthreads();
    if (tid < TILE) {
        float s = 0.0f;
#pragma unroll
        for (int w = 0; w < 8; ++w) {
            s += red[w * 32 + tid];
            s += red[w * 32 + 16 + tid];
        }
        spart[(size_t)mchk * D_FF + n0 + tid] = s;
    }
}

__global__ void scores_reduce(const float* __restrict__ spart,
                              float* __restrict__ scores) {
    int f = blockIdx.x * blockDim.x + threadIdx.x;
    if (f < D_FF) {
        float s = 0.0f;
#pragma unroll
        for (int c = 0; c < 16; ++c) s += spart[(size_t)c * D_FF + f];
        scores[f] = s;   // mean scale omitted: rank-invariant
    }
}

// exact-rank top-k (ranks are a permutation -> race-free, deterministic)
__global__ __launch_bounds__(1024)
void topk_select(const float* __restrict__ scores, int* __restrict__ selected) {
    __shared__ float s[D_FF];
    const int tid = threadIdx.x;
    for (int i = tid; i < D_FF; i += 1024) s[i] = scores[i];
    __syncthreads();
    for (int f = tid; f < D_FF; f += 1024) {
        const float v = s[f];
        int r = 0;
        for (int g = 0; g < D_FF; ++g) {
            float u = s[g];
            r += (u > v) || (u == v && g < f);
        }
        if (r < K_SEL) selected[r] = f;
    }
}

// inter_sel = relu(x @ w1[selected]^T + b1[selected]); w1 rows gathered by TDM
__global__ __launch_bounds__(256)
void lin1_gemm(const __bf16* __restrict__ xb,
               const __bf16* __restrict__ w1b,
               const float*  __restrict__ b1,
               const int*    __restrict__ sel,
               __bf16* __restrict__ inter) {
    __shared__ __align__(16) __bf16 Bs[TILE * D_MODEL];   // 32KB

    const int tid   = threadIdx.x;
    const int ntile = blockIdx.x;           // 0..127
    const int mchk  = blockIdx.y;           // 0..15
    const int n0    = ntile * TILE;

    // gather the 16 selected w1 rows into LDS with one TDM gather-mode op
    if (tid < 32) {
        tdm_gather_rows16((unsigned int)(uintptr_t)&Bs[0],
                          w1b, D_MODEL, D_FF, sel + n0);
        __builtin_amdgcn_s_wait_tensorcnt(0);
    }
    __syncthreads();

    const int wave = tid >> 5, lane = tid & 31;
    const int col   = lane & 15;
    const int rbase = (lane >> 4) << 3;
    const float bias = b1[sel[n0 + col]];

    for (int g = 0; g < 2; ++g) {
        const int mt = mchk * 64 + wave * 8 + g * 4;
        v8f acc[4] = {};
        for (int ks = 0; ks < D_MODEL; ks += KSTEP) {
            v16bf b = load_b_frag(Bs, D_MODEL, ks, lane);
            __builtin_prefetch(xb + (size_t)(mt * TILE) * D_MODEL + ks + 256, 0, 1);
#pragma unroll
            for (int i = 0; i < 4; ++i) {
                v16bf a = load_a_frag(xb, D_MODEL, (mt + i) * TILE, ks, lane);
                acc[i] = wmma_bf16(a, b, acc[i]);
            }
        }
#pragma unroll
        for (int i = 0; i < 4; ++i) {
            const int mrow = (mt + i) * TILE + rbase;
#pragma unroll
            for (int v = 0; v < 8; ++v) {
                float val = fmaxf(acc[i][v] + bias, 0.0f);
                inter[(size_t)(mrow + v) * K_SEL + n0 + col] = (__bf16)val;
            }
        }
    }
}

// out = inter_sel @ w2[:, selected]^T + b2 (element gather kept on LDS path)
__global__ __launch_bounds__(256)
void lin2_gemm(const __bf16* __restrict__ inter,
               const __bf16* __restrict__ w2b,
               const float*  __restrict__ b2,
               const int*    __restrict__ sel,
               float* __restrict__ out) {
    __shared__ __align__(16) __bf16 Ws[TILE * K_SEL];     // 64KB
    __shared__ int selS[K_SEL];                           // 8KB

    const int tid   = threadIdx.x;
    const int ntile = blockIdx.x;           // 0..63
    const int mchk  = blockIdx.y;           // 0..15
    const int n0    = ntile * TILE;

    for (int j = tid; j < K_SEL; j += 256) selS[j] = sel[j];
    __syncthreads();
    for (int i = tid; i < TILE * K_SEL; i += 256) {
        int d = i >> 11, j = i & (K_SEL - 1);
        Ws[d * K_SEL + j] = w2b[(size_t)(n0 + d) * D_FF + selS[j]];
    }
    __syncthreads();

    const int wave = tid >> 5, lane = tid & 31;
    const int col   = lane & 15;
    const int rbase = (lane >> 4) << 3;
    const float bias = b2[n0 + col];

    for (int g = 0; g < 2; ++g) {
        const int mt = mchk * 64 + wave * 8 + g * 4;
        v8f acc[4] = {};
        for (int ks = 0; ks < K_SEL; ks += KSTEP) {
            v16bf b = load_b_frag(Ws, K_SEL, ks, lane);
            __builtin_prefetch(inter + (size_t)(mt * TILE) * K_SEL + ks + 256, 0, 1);
#pragma unroll
            for (int i = 0; i < 4; ++i) {
                v16bf a = load_a_frag(inter, K_SEL, (mt + i) * TILE, ks, lane);
                acc[i] = wmma_bf16(a, b, acc[i]);
            }
        }
#pragma unroll
        for (int i = 0; i < 4; ++i) {
            const int mrow = (mt + i) * TILE + rbase;
#pragma unroll
            for (int v = 0; v < 8; ++v)
                out[(size_t)(mrow + v) * D_MODEL + n0 + col] = acc[i][v] + bias;
        }
    }
}

// ---------------- launch ----------------
extern "C" void kernel_launch(void* const* d_in, const int* in_sizes, int n_in,
                              void* d_out, int out_size, void* d_ws, size_t ws_size,
                              hipStream_t stream) {
    const float* x     = (const float*)d_in[0];
    const float* w1    = (const float*)d_in[1];
    const float* b1    = (const float*)d_in[2];
    const float* w2    = (const float*)d_in[3];
    const float* b2    = (const float*)d_in[4];
    const float* w_est = (const float*)d_in[5];
    const float* b_est = (const float*)d_in[6];
    float* out = (float*)d_out;

    char* ws = (char*)d_ws;
    __bf16* xb    = (__bf16*)(ws + WS_XB);
    __bf16* w1b   = (__bf16*)(ws + WS_W1B);
    __bf16* westb = (__bf16*)(ws + WS_WESTB);
    __bf16* w2b   = (__bf16*)(ws + WS_W2B);
    __bf16* inter = (__bf16*)(ws + WS_INTER);
    float*  spart = (float*)(ws + WS_SPART);
    float*  score = (float*)(ws + WS_SCORE);
    int*    sel   = (int*)(ws + WS_SEL);

    {
        int n4 = (M_TOT * D_MODEL) / 4;
        cvt_f32_bf16<<<(n4 + 255) / 256, 256, 0, stream>>>(x, xb, n4);
        n4 = (D_FF * D_MODEL) / 4;
        cvt_f32_bf16<<<(n4 + 255) / 256, 256, 0, stream>>>(w1, w1b, n4);
        cvt_f32_bf16<<<(n4 + 255) / 256, 256, 0, stream>>>(w_est, westb, n4);
        cvt_f32_bf16<<<(n4 + 255) / 256, 256, 0, stream>>>(w2, w2b, n4);
    }
    scores_gemm<<<dim3(D_FF / TILE, 16), 256, 0, stream>>>(xb, westb, b_est, spart);
    scores_reduce<<<(D_FF + 255) / 256, 256, 0, stream>>>(spart, score);
    topk_select<<<1, 1024, 0, stream>>>(score, sel);
    lin1_gemm<<<dim3(K_SEL / TILE, 16), 256, 0, stream>>>(xb, w1b, b1, sel, inter);
    lin2_gemm<<<dim3(D_MODEL / TILE, 16), 256, 0, stream>>>(inter, w2b, b2, sel, out);
    (void)in_sizes; (void)n_in; (void)out_size; (void)ws_size;
}
